// Block_8632884265124
// MI455X (gfx1250) — matrix operand order repaired and checked by reading, not compiled, not run on hardware
//
#include <hip/hip_runtime.h>
#include <math.h>
#include <stdint.h>

// ---------------------------------------------------------------------------
// Types for CDNA5 WMMA (wave32): bf16 A/B fragments, f32 accumulators.
// ---------------------------------------------------------------------------
typedef __attribute__((ext_vector_type(16))) __bf16 v16bf;
typedef __attribute__((ext_vector_type(8)))  __bf16 v8bf;
typedef __attribute__((ext_vector_type(8)))  float  v8f;
typedef __attribute__((ext_vector_type(4)))  unsigned int u32x4;
typedef __attribute__((ext_vector_type(4)))  int i32x4;
typedef __attribute__((ext_vector_type(8)))  int i32x8;

#define D_MODEL 1024
#define SEQ_T   2048
#define NBATCH  2
#define MROWS   (NBATCH * SEQ_T)   // 4096
#define NHEAD   16
#define HDIM    64

#if defined(__has_builtin)
#if __has_builtin(__builtin_amdgcn_tensor_load_to_lds) && \
    __has_builtin(__builtin_amdgcn_s_wait_tensorcnt)
#define HAVE_TDM 1
#endif
#endif
#ifndef HAVE_TDM
#define HAVE_TDM 0
#endif

__device__ __forceinline__ v8f wmma_bf16(v16bf a, v16bf b, v8f c) {
    // D = A(16x32 bf16) x B(32x16 bf16) + C(16x16 f32)
    return __builtin_amdgcn_wmma_f32_16x16x32_bf16(
        false, a, false, b, (short)0, c, false, false);
}

// A-matrix fragment (16x32 bf16): per ISA 7.12.2 the per-lane K window is
// interleaved: elements {o..o+7, o+16..o+23} with o = (lane>=16 ? 8 : 0).
__device__ __forceinline__ v16bf load_fragA(const __bf16* p) {
    v8bf lo = *(const v8bf*)(p);
    v8bf hi = *(const v8bf*)(p + 16);
    v16bf r;
#pragma unroll
    for (int i = 0; i < 8; ++i) { r[i] = lo[i]; r[i + 8] = hi[i]; }
    return r;
}

// B-matrix fragment (32x16 bf16): per-lane K window is interleaved like A:
// {ob..ob+7, ob+16..ob+23} with ob = (lane>=16 ? 8 : 0); memory contiguous along K.
__device__ __forceinline__ v16bf load_fragB(const __bf16* p) {
    v8bf lo = *(const v8bf*)(p);
    v8bf hi = *(const v8bf*)(p + 16);
    v16bf r;
#pragma unroll
    for (int i = 0; i < 8; ++i) { r[i] = lo[i]; r[i + 8] = hi[i]; }
    return r;
}

__device__ __forceinline__ v8f zero8() {
    v8f z;
#pragma unroll
    for (int i = 0; i < 8; ++i) z[i] = 0.0f;
    return z;
}

#if HAVE_TDM
// ---------------------------------------------------------------------------
// Tensor Data Mover: 2D bf16 tile (tile_k x tile_rows) from global (row
// stride = row_stride_elems) into LDS at lds_byte, with LDS padding of
// pad_amount (code: n -> (n+1) DWORDs) every 2^(pad_code+1) DWORDs so the
// staged tile matches the padded LDS layout the WMMA fragment loads expect.
// Descriptor layout per CDNA5 ISA.
// ---------------------------------------------------------------------------
__device__ __forceinline__ void tdm_load_2d_bf16(const __bf16* gptr,
                                                 unsigned int lds_byte,
                                                 int tile_k, int tile_rows,
                                                 int row_stride_elems,
                                                 int pad_interval_code,
                                                 int pad_amount_code) {
    uint64_t ga = (uint64_t)(uintptr_t)gptr;
    u32x4 g0;
    g0[0] = 1u;                                   // count=1, user descriptor
    g0[1] = lds_byte;                             // lds_addr [63:32]
    g0[2] = (unsigned int)ga;                     // global_addr [95:64]
    g0[3] = (unsigned int)((ga >> 32) & 0x01FFFFFFu) | (2u << 30);  // +type=2
    i32x8 g1;
    g1[0] = (1 << 16)                             // data_size = 2 bytes
          | (1 << 20)                             // pad_enable
          | (pad_interval_code << 22)
          | (pad_amount_code << 25);
    g1[1] = (tile_k & 0xFFFF) << 16;              // tensor_dim0 lo16 @ 63:48
    g1[2] = ((tile_k >> 16) & 0xFFFF)             // tensor_dim0 hi16
          | ((tile_rows & 0xFFFF) << 16);         // tensor_dim1 lo16
    g1[3] = ((tile_rows >> 16) & 0xFFFF)          // tensor_dim1 hi16
          | ((tile_k & 0xFFFF) << 16);            // tile_dim0 @ 127:112
    g1[4] = (tile_rows & 0xFFFF);                 // tile_dim1 ; tile_dim2=0
    g1[5] = row_stride_elems;                     // tensor_dim0_stride lo32
    g1[6] = 0;                                    // stride hi / dim1_stride
    g1[7] = 0;
    i32x4 z4 = {0, 0, 0, 0};
#if __clang_major__ >= 23
    i32x8 z8 = {0, 0, 0, 0, 0, 0, 0, 0};
    __builtin_amdgcn_tensor_load_to_lds(g0, g1, z4, z4, z8, 0);
#else
    __builtin_amdgcn_tensor_load_to_lds(g0, g1, z4, z4, 0);
#endif
}
#endif  // HAVE_TDM

// ---------------------------------------------------------------------------
// fp32 -> bf16 elementwise convert (weights + activations staging)
// ---------------------------------------------------------------------------
__global__ void cvt_f32_bf16(const float* __restrict__ src,
                             __bf16* __restrict__ dst, int n) {
    int i = (blockIdx.x * 256 + threadIdx.x) * 4;
    if (i < n) {
        float4 v = *(const float4*)(src + i);
        dst[i + 0] = (__bf16)v.x;
        dst[i + 1] = (__bf16)v.y;
        dst[i + 2] = (__bf16)v.z;
        dst[i + 3] = (__bf16)v.w;
    }
}

// ---------------------------------------------------------------------------
// LayerNorm over C=1024: one block (256 threads) per row, bf16 output.
// ---------------------------------------------------------------------------
__global__ __launch_bounds__(256)
void ln_kernel(const float* __restrict__ x, const float* __restrict__ g,
               const float* __restrict__ b, __bf16* __restrict__ out) {
    __shared__ float ssum[8];
    __shared__ float ssq[8];
    int row = blockIdx.x;
    int tid = threadIdx.x;
    const float4 v = ((const float4*)(x + (size_t)row * D_MODEL))[tid];
    float s = v.x + v.y + v.z + v.w;
    float q = v.x * v.x + v.y * v.y + v.z * v.z + v.w * v.w;
#pragma unroll
    for (int m = 16; m >= 1; m >>= 1) {
        s += __shfl_xor(s, m);
        q += __shfl_xor(q, m);
    }
    if ((tid & 31) == 0) { ssum[tid >> 5] = s; ssq[tid >> 5] = q; }
    __syncthreads();
    float ts = 0.0f, tq = 0.0f;
#pragma unroll
    for (int i = 0; i < 8; ++i) { ts += ssum[i]; tq += ssq[i]; }
    const float inv = 1.0f / (float)D_MODEL;
    float mu = ts * inv;
    float var = tq * inv - mu * mu;
    float rstd = rsqrtf(var + 1e-5f);
    float4 gv = ((const float4*)g)[tid];
    float4 bv = ((const float4*)b)[tid];
    __bf16* o = out + (size_t)row * D_MODEL + tid * 4;
    o[0] = (__bf16)((v.x - mu) * rstd * gv.x + bv.x);
    o[1] = (__bf16)((v.y - mu) * rstd * gv.y + bv.y);
    o[2] = (__bf16)((v.z - mu) * rstd * gv.z + bv.z);
    o[3] = (__bf16)((v.w - mu) * rstd * gv.w + bv.w);
}

// ---------------------------------------------------------------------------
// bf16 WMMA GEMM: out[M,N] = A[M,K] @ W[K,N] + bias (+ GELU) (+ residual)
// Block tile 128x128, 8 waves, each wave 64x32 (4x2 WMMA tiles).
// A tile staged by the Tensor Data Mover (wave 0 issues the descriptor, LDS
// padding in hardware); B tile staged manually with an in-flight transpose.
// ---------------------------------------------------------------------------
template <int OUT_BF16, int GELU, int RES>
__global__ __launch_bounds__(256)
void gemm_wmma(const __bf16* __restrict__ A, const __bf16* __restrict__ W,
               const float* __restrict__ bias, const float* __restrict__ res,
               void* __restrict__ out, int M, int N, int K) {
    __shared__ __bf16 sA[128 * 72];  // [row][k] padded (72 = 64 + 8)
    __shared__ __bf16 sB[128 * 72];  // [n][k] transposed, padded

    const int tid = threadIdx.x;
    const int w = tid >> 5;
    const int lane = tid & 31;
    const int idx = lane & 15;
    const int hi = lane >> 4;
    const int o = hi * 8;    // A-frag K offset (interleaved)
    const int ob = hi * 8;   // B-frag K offset (interleaved)

    const int waveM = w >> 2;  // 0..1  -> 64 rows each
    const int waveN = w & 3;   // 0..3  -> 32 cols each
    const int rowBase = blockIdx.y * 128;
    const int colBase = blockIdx.x * 128;

#if HAVE_TDM
    // Low 32 bits of the generic shared address = wave-relative LDS byte
    // address (flat->LDS aperture truncation, ISA 10.2).
    const unsigned int sA_lds = (unsigned int)(uintptr_t)(&sA[0]);
#endif

    v8f acc[4][2];
#pragma unroll
    for (int a = 0; a < 4; ++a)
#pragma unroll
        for (int b2 = 0; b2 < 2; ++b2) acc[a][b2] = zero8();

    for (int k0 = 0; k0 < K; k0 += 64) {
        // ---- Stage A tile 128x64 ----
#if HAVE_TDM
        if (w == 0) {
            // pad_interval code 4 -> every 32 DWORDs (= one 64-elem bf16 row)
            // pad_amount  code 3 -> 4 DWORDs (= 8 bf16)  => 72-elem rows.
            tdm_load_2d_bf16(A + (size_t)rowBase * K + k0, sA_lds,
                             /*tile_k=*/64, /*tile_rows=*/128,
                             /*stride=*/K, /*pad_i=*/4, /*pad_a=*/3);
        }
#else
#pragma unroll
        for (int it = 0; it < 4; ++it) {
            int v = tid + it * 256;           // 1024 vectors of 8 bf16
            int r = v >> 3;
            int kc = (v & 7) << 3;
            v8bf d = *(const v8bf*)(A + (size_t)(rowBase + r) * K + k0 + kc);
            *(v8bf*)(sA + r * 72 + kc) = d;
        }
#endif
        // ---- Stage B tile 64x128 transposed into [n][k] ----
#pragma unroll
        for (int it = 0; it < 4; ++it) {
            int v = tid + it * 256;           // 1024 vectors of 8 bf16
            int kr = v >> 4;
            int nc = (v & 15) << 3;
            v8bf d = *(const v8bf*)(W + (size_t)(k0 + kr) * N + colBase + nc);
#pragma unroll
            for (int t = 0; t < 8; ++t) sB[(nc + t) * 72 + kr] = d[t];
        }
        // ---- Prefetch next K-step tiles into L2 (global_prefetch_b8) ----
        if (k0 + 64 < K) {
            __builtin_prefetch(A + (size_t)(rowBase + (tid >> 1)) * K +
                                   (k0 + 64) + (tid & 1) * 32, 0, 1);
            __builtin_prefetch(W + (size_t)(k0 + 64 + (tid >> 2)) * N +
                                   colBase + (tid & 3) * 32, 0, 1);
        }
#if HAVE_TDM
        if (w == 0) __builtin_amdgcn_s_wait_tensorcnt(0);
#endif
        __syncthreads();

#pragma unroll
        for (int c = 0; c < 2; ++c) {
            v16bf af[4];
#pragma unroll
            for (int mi = 0; mi < 4; ++mi)
                af[mi] = load_fragA(sA + (waveM * 64 + mi * 16 + idx) * 72 + c * 32 + o);
            v16bf bfr[2];
#pragma unroll
            for (int ni = 0; ni < 2; ++ni)
                bfr[ni] = load_fragB(sB + (waveN * 32 + ni * 16 + idx) * 72 + c * 32 + ob);
#pragma unroll
            for (int mi = 0; mi < 4; ++mi)
#pragma unroll
                for (int ni = 0; ni < 2; ++ni)
                    acc[mi][ni] = wmma_bf16(af[mi], bfr[ni], acc[mi][ni]);
        }
        __syncthreads();
    }

    // Epilogue: C/D layout: element i, lane -> row = i + 8*hi, col = idx.
#pragma unroll
    for (int mi = 0; mi < 4; ++mi) {
#pragma unroll
        for (int ni = 0; ni < 2; ++ni) {
            int col = colBase + waveN * 32 + ni * 16 + idx;
            float bv = bias[col];
#pragma unroll
            for (int i = 0; i < 8; ++i) {
                int row = rowBase + waveM * 64 + mi * 16 + i + o;
                float val = acc[mi][ni][i] + bv;
                if (GELU) val = 0.5f * val * (1.0f + erff(val * 0.70710678118f));
                size_t off = (size_t)row * N + col;
                if (RES) val += res[off];
                if (OUT_BF16) ((__bf16*)out)[off] = (__bf16)val;
                else          ((float*)out)[off] = val;
            }
        }
    }
}

// ---------------------------------------------------------------------------
// Causal flash attention with WMMA.
// Grid: (T/128, NHEAD, B). Block: 256 threads = 8 waves; each wave owns 16
// query rows. S = Q K^T via wmma; online softmax (row stats via shfl across
// the 16-lane half, matching the C/D row layout); P re-staged through LDS as
// A fragments; O += P V via wmma.
// qkv layout: [B*T, 3C] bf16, head h at cols h*64 (+0 q, +1024 k, +2048 v).
// ---------------------------------------------------------------------------
__global__ __launch_bounds__(256)
void attn_kernel(const __bf16* __restrict__ qkv, __bf16* __restrict__ y) {
    __shared__ __bf16 sK[64 * 72];        // [key][hd]
    __shared__ __bf16 sVt[64 * 72];       // [hd][key]
    __shared__ __bf16 sP[8 * 16 * 72];    // per-wave 16x64 P tile

    const int qb = blockIdx.x * 128;
    const int h = blockIdx.y;
    const int b = blockIdx.z;
    const int tid = threadIdx.x;
    const int w = tid >> 5;
    const int lane = tid & 31;
    const int idx = lane & 15;
    const int hiH = lane >> 4;
    const int o = hiH * 8;
    const int ob = hiH * 8;

    const __bf16* qbase = qkv + (size_t)(b * SEQ_T) * 3072 + h * HDIM;
    const __bf16* kbase = qbase + 1024;
    const __bf16* vbase = qbase + 2048;

    // Q fragments for this wave's 16 rows (A-frag: lane idx -> row).
    const int qrow = qb + w * 16 + idx;
    v16bf qf[2];
#pragma unroll
    for (int c = 0; c < 2; ++c)
        qf[c] = load_fragA(qbase + (size_t)qrow * 3072 + c * 32 + o);

    v8f Ot[4];
#pragma unroll
    for (int t = 0; t < 4; ++t) Ot[t] = zero8();
    float m_i[8], l_i[8];
#pragma unroll
    for (int i = 0; i < 8; ++i) { m_i[i] = -1e30f; l_i[i] = 0.0f; }

    const int nkt = (qb >> 6) + 2;  // key tiles covering keys <= qb+127
    for (int kt = 0; kt < nkt; ++kt) {
        const int kb = kt * 64;
        // Cooperative stage: K natural [key][hd], V transposed [hd][key].
#pragma unroll
        for (int it = 0; it < 2; ++it) {
            int v = tid + it * 256;          // 512 vectors of 8 bf16
            int key = v >> 3;
            int dc = (v & 7) << 3;
            v8bf dk = *(const v8bf*)(kbase + (size_t)(kb + key) * 3072 + dc);
            *(v8bf*)(sK + key * 72 + dc) = dk;
            v8bf dv = *(const v8bf*)(vbase + (size_t)(kb + key) * 3072 + dc);
#pragma unroll
            for (int t = 0; t < 8; ++t) sVt[(dc + t) * 72 + key] = dv[t];
        }
        // Prefetch next key tile (K and V rows) while computing this one.
        if (kt + 1 < nkt) {
            __builtin_prefetch(kbase + (size_t)(kb + 64 + (tid >> 2)) * 3072 +
                                   (tid & 3) * 16, 0, 1);
            __builtin_prefetch(vbase + (size_t)(kb + 64 + (tid >> 2)) * 3072 +
                                   (tid & 3) * 16, 0, 1);
        }
        __syncthreads();

        // S tiles: 16 rows x 64 keys (4 tiles of 16x16).
        v8f s[4];
#pragma unroll
        for (int j = 0; j < 4; ++j) {
            s[j] = zero8();
#pragma unroll
            for (int c = 0; c < 2; ++c) {
                v16bf kf = load_fragB(sK + (j * 16 + idx) * 72 + c * 32 + ob);
                s[j] = wmma_bf16(qf[c], kf, s[j]);
            }
        }
        // Scale + causal mask. Element (i,lane) of tile j:
        //   q = qb + w*16 + i + o ;  k = kb + j*16 + idx
#pragma unroll
        for (int j = 0; j < 4; ++j) {
            int kcol = kb + j * 16 + idx;
#pragma unroll
            for (int i = 0; i < 8; ++i) {
                int qg = qb + w * 16 + i + o;
                float sv = s[j][i] * 0.125f;  // 1/sqrt(64)
                if (kcol > qg) sv = -1e30f;
                s[j][i] = sv;
            }
        }
        // Row max across 64 keys (4 tiles + 16-lane half reduce).
        float mnew[8];
#pragma unroll
        for (int i = 0; i < 8; ++i) {
            float mx = s[0][i];
#pragma unroll
            for (int j = 1; j < 4; ++j) mx = fmaxf(mx, s[j][i]);
#pragma unroll
            for (int m = 8; m >= 1; m >>= 1) mx = fmaxf(mx, __shfl_xor(mx, m));
            mnew[i] = fmaxf(m_i[i], mx);
        }
        // P = exp(S - mnew), row sums, rescale O.
        float rs[8];
#pragma unroll
        for (int i = 0; i < 8; ++i) rs[i] = 0.0f;
#pragma unroll
        for (int j = 0; j < 4; ++j)
#pragma unroll
            for (int i = 0; i < 8; ++i) {
                float p = __expf(s[j][i] - mnew[i]);
                s[j][i] = p;
                rs[i] += p;
            }
#pragma unroll
        for (int i = 0; i < 8; ++i) {
#pragma unroll
            for (int m = 8; m >= 1; m >>= 1) rs[i] += __shfl_xor(rs[i], m);
            float alpha = __expf(m_i[i] - mnew[i]);
            l_i[i] = l_i[i] * alpha + rs[i];
            m_i[i] = mnew[i];
#pragma unroll
            for (int t = 0; t < 4; ++t) Ot[t][i] *= alpha;
        }
        // Store P (bf16) to this wave's LDS region, reload as A-fragments.
        __bf16* pw = sP + w * (16 * 72);
#pragma unroll
        for (int j = 0; j < 4; ++j)
#pragma unroll
            for (int i = 0; i < 8; ++i)
                pw[(i + o) * 72 + j * 16 + idx] = (__bf16)s[j][i];
        // O += P @ V  (V as B-frag from transposed tile)
#pragma unroll
        for (int c = 0; c < 2; ++c) {
            v16bf pf = load_fragA(pw + idx * 72 + c * 32 + o);
#pragma unroll
            for (int t = 0; t < 4; ++t) {
                v16bf vf = load_fragB(sVt + (t * 16 + idx) * 72 + c * 32 + ob);
                Ot[t] = wmma_bf16(pf, vf, Ot[t]);
            }
        }
        __syncthreads();  // before next tile overwrites sK/sVt
    }

    // Normalize and write y[b, q, h*64 + d] (bf16).
#pragma unroll
    for (int t = 0; t < 4; ++t)
#pragma unroll
        for (int i = 0; i < 8; ++i) {
            int qg = qb + w * 16 + i + o;
            float ov = Ot[t][i] / l_i[i];
            y[(size_t)(b * SEQ_T + qg) * D_MODEL + h * HDIM + t * 16 + idx] =
                (__bf16)ov;
        }
}

// ---------------------------------------------------------------------------
// Host orchestration
// ---------------------------------------------------------------------------
extern "C" void kernel_launch(void* const* d_in, const int* in_sizes, int n_in,
                              void* d_out, int out_size, void* d_ws, size_t ws_size,
                              hipStream_t stream) {
    const float* x      = (const float*)d_in[0];
    const float* ln1_g  = (const float*)d_in[1];
    const float* ln1_b  = (const float*)d_in[2];
    const float* w_attn = (const float*)d_in[3];
    const float* b_attn = (const float*)d_in[4];
    const float* w_ap   = (const float*)d_in[5];
    const float* b_ap   = (const float*)d_in[6];
    const float* ln2_g  = (const float*)d_in[7];
    const float* ln2_b  = (const float*)d_in[8];
    const float* w_fc   = (const float*)d_in[9];
    const float* b_fc   = (const float*)d_in[10];
    const float* w_mp   = (const float*)d_in[11];
    const float* b_mp   = (const float*)d_in[12];

    char* ws = (char*)d_ws;
    size_t cur = 0;
    auto alloc = [&](size_t bytes) -> void* {
        void* p = ws + cur;
        cur += (bytes + 255) & ~(size_t)255;
        return p;
    };
    __bf16* w_attn_bf = (__bf16*)alloc((size_t)1024 * 3072 * 2);
    __bf16* w_ap_bf   = (__bf16*)alloc((size_t)1024 * 1024 * 2);
    __bf16* w_fc_bf   = (__bf16*)alloc((size_t)1024 * 4096 * 2);
    __bf16* w_mp_bf   = (__bf16*)alloc((size_t)4096 * 1024 * 2);
    __bf16* ln1_bf    = (__bf16*)alloc((size_t)MROWS * 1024 * 2);
    __bf16* qkv_bf    = (__bf16*)alloc((size_t)MROWS * 3072 * 2);
    __bf16* y_bf      = (__bf16*)alloc((size_t)MROWS * 1024 * 2);
    float*  x1_f      = (float*) alloc((size_t)MROWS * 1024 * 4);
    __bf16* ln2_bf    = (__bf16*)alloc((size_t)MROWS * 1024 * 2);
    __bf16* h_bf      = (__bf16*)alloc((size_t)MROWS * 4096 * 2);

    // Convert weights to bf16.
    cvt_f32_bf16<<<(1024 * 3072) / 1024, 256, 0, stream>>>(w_attn, w_attn_bf, 1024 * 3072);
    cvt_f32_bf16<<<(1024 * 1024) / 1024, 256, 0, stream>>>(w_ap,   w_ap_bf,   1024 * 1024);
    cvt_f32_bf16<<<(1024 * 4096) / 1024, 256, 0, stream>>>(w_fc,   w_fc_bf,   1024 * 4096);
    cvt_f32_bf16<<<(4096 * 1024) / 1024, 256, 0, stream>>>(w_mp,   w_mp_bf,   4096 * 1024);

    // 1) ln1
    ln_kernel<<<MROWS, 256, 0, stream>>>(x, ln1_g, ln1_b, ln1_bf);
    // 2) qkv = ln1 @ w_attn + b_attn   -> bf16
    gemm_wmma<1, 0, 0><<<dim3(3072 / 128, MROWS / 128), 256, 0, stream>>>(
        ln1_bf, w_attn_bf, b_attn, nullptr, qkv_bf, MROWS, 3072, 1024);
    // 3) causal MHA -> y (bf16)
    attn_kernel<<<dim3(SEQ_T / 128, NHEAD, NBATCH), 256, 0, stream>>>(qkv_bf, y_bf);
    // 4) x1 = x + y @ w_attn_proj + b  -> f32
    gemm_wmma<0, 0, 1><<<dim3(1024 / 128, MROWS / 128), 256, 0, stream>>>(
        y_bf, w_ap_bf, b_ap, x, x1_f, MROWS, 1024, 1024);
    // 5) ln2
    ln_kernel<<<MROWS, 256, 0, stream>>>(x1_f, ln2_g, ln2_b, ln2_bf);
    // 6) h = gelu(ln2 @ w_fc + b_fc)   -> bf16
    gemm_wmma<1, 1, 0><<<dim3(4096 / 128, MROWS / 128), 256, 0, stream>>>(
        ln2_bf, w_fc_bf, b_fc, nullptr, h_bf, MROWS, 4096, 1024);
    // 7) out = x1 + h @ w_mlp_proj + b -> f32 (d_out)
    gemm_wmma<0, 0, 1><<<dim3(1024 / 128, MROWS / 128), 256, 0, stream>>>(
        h_bf, w_mp_bf, b_mp, x1_f, d_out, MROWS, 1024, 4096);

    (void)in_sizes; (void)n_in; (void)out_size; (void)ws_size;
}
